// c2vRNNModel_64836826300637
// MI455X (gfx1250) — matrix operand order repaired and checked by reading, not compiled, not run on hardware
//
#include <hip/hip_runtime.h>
#include <math.h>

// ---------------- model constants ----------------
#define Bc     16
#define Tc     128
#define Lc     100
#define INDIM  100
#define EMBc   100
#define FEATc  400      // 3*EMB + INPUT_DIM
#define KP     416      // FEAT padded to multiple of 32
#define XD     400      // INPUT_DIM + 3*L  (x row length)
#define HIDc   128
#define G4     512      // 4*HID
#define RINP   512      // rnn input (500) padded
#define KLSTM  640      // 500 + 128 padded to 32
#define OUTc   50
#define VOCABc 10002

typedef __attribute__((ext_vector_type(16))) __bf16 v16bf;
typedef __attribute__((ext_vector_type(8)))  float  v8f;

union BFrag { v16bf v; uint4 q[2]; };

// A-matrix 16x32 bf16 fragment (ISA 7.12.2): lane m = lane&15, hk = lane>>4.
// regs0-3 hold K = hk*8 + 0..7, regs4-7 hold K = 16 + hk*8 + 0..7.
// `row` points at element [m][kc*32] of a row-major A (row stride aligned 16B).
__device__ __forceinline__ v16bf load_a16(const __bf16* row, int hk) {
    BFrag u;
    u.q[0] = *reinterpret_cast<const uint4*>(row + hk * 8);
    u.q[1] = *reinterpret_cast<const uint4*>(row + 16 + hk * 8);
    return u.v;
}
// B-matrix 32x16 bf16 fragment: lane holds column n = lane&15; its 16 elements
// are K = hk*8 + 0..7 and 16 + hk*8 + 0..7 (B stored transposed [n][k]).
__device__ __forceinline__ v16bf load_b16(const __bf16* rowT, int hk) {
    BFrag u;
    u.q[0] = *reinterpret_cast<const uint4*>(rowT + hk * 8);
    u.q[1] = *reinterpret_cast<const uint4*>(rowT + 16 + hk * 8);
    return u.v;
}
__device__ __forceinline__ v8f wmma_bf16(v16bf a, v16bf b, v8f c) {
    return __builtin_amdgcn_wmma_f32_16x16x32_bf16(false, a, false, b,
                                                   (short)0, c, false, false);
}
__device__ __forceinline__ float sigf(float x) { return 1.0f / (1.0f + expf(-x)); }

// ---------------- prep kernels: bf16 transposed, zero-padded weights --------
__global__ void prep_wt(const float* __restrict__ Wt, __bf16* __restrict__ WtT) {
    int i = blockIdx.x * blockDim.x + threadIdx.x;              // [n][k], 416x416
    if (i >= KP * KP) return;
    int n = i / KP, k = i % KP;
    float v = (n < FEATc && k < FEATc) ? Wt[k * FEATc + n] : 0.0f;
    WtT[i] = (__bf16)v;
}
__global__ void prep_wcat(const float* __restrict__ Wih, const float* __restrict__ Whh,
                          __bf16* __restrict__ WcT) {
    int i = blockIdx.x * blockDim.x + threadIdx.x;              // [g][k], 512x640
    if (i >= G4 * KLSTM) return;
    int g = i / KLSTM, k = i % KLSTM;
    float v = 0.0f;
    if (k < 500)       v = Wih[g * 500 + k];
    else if (k < 628)  v = Whh[g * HIDc + (k - 500)];
    WcT[i] = (__bf16)v;
}
__global__ void prep_wfc(const float* __restrict__ Wfc, __bf16* __restrict__ WfT) {
    int i = blockIdx.x * blockDim.x + threadIdx.x;              // [o][h], 64x128
    if (i >= 64 * HIDc) return;
    int o = i / HIDc, h = i % HIDc;
    float v = (o < OUTc) ? Wfc[h * OUTc + o] : 0.0f;
    WfT[i] = (__bf16)v;
}
__global__ void prep_bias(const float* __restrict__ bih, const float* __restrict__ bhh,
                          float* __restrict__ bsum) {
    int i = blockIdx.x * blockDim.x + threadIdx.x;
    if (i < G4) bsum[i] = bih[i] + bhh[i];
}

// ---------------- kernel 1: ctx[b,l,t] = tanh(full@Wt + bt) @ Wa ------------
// One block per (b,t). full (100x400) built in LDS as bf16 in two 64-row
// phases. Loop nt-outer with the 13 B fragments held in registers (104 VGPRs)
// and reused across 4 M-tiles; epilogue reduces the 16 columns per row with a
// shfl_xor butterfly before a single ds_add_f32 per row.
// launch_bounds(256,1): 8 waves, one block resident -> no VGPR cap, no spills.
__global__ void __launch_bounds__(256, 1)
attn_score_kernel(const float* __restrict__ x,
                  const float* __restrict__ embN,
                  const float* __restrict__ embP,
                  const __bf16* __restrict__ WtT,
                  const float* __restrict__ bt_vec,
                  const float* __restrict__ Wa,
                  float* __restrict__ ctx) {
    __shared__ __bf16 Atile[64 * KP];          // 53 KB
    __shared__ int   sIdx[Lc], pIdx[Lc], eIdx[Lc];
    __shared__ float ctx_acc[128];

    const int bt = blockIdx.x, b = bt / Tc, t = bt % Tc;
    const int tid = threadIdx.x;
    const float* xrow = x + (size_t)bt * XD;

    for (int i = tid; i < 3 * Lc; i += 256) {
        int iv = (int)xrow[INDIM + i];
        iv = iv < 0 ? 0 : (iv >= VOCABc ? VOCABc - 1 : iv);
        int l = i / 3, c = i % 3;
        if (c == 0) sIdx[l] = iv; else if (c == 1) pIdx[l] = iv; else eIdx[l] = iv;
    }
    if (tid < 128) ctx_acc[tid] = 0.0f;
    __syncthreads();

    const int lane = tid & 31, wv = tid >> 5;
    const int hk = lane >> 4, nlo = lane & 15;

    for (int phase = 0; phase < 2; ++phase) {
        // fill 64 rows (l = phase*64 + r) of bf16 full-tile, zero padded
        for (int e = tid; e < 64 * KP; e += 256) {
            int r = e / KP, k = e % KP, l = phase * 64 + r;
            float v = 0.0f;
            if (l < Lc && k < FEATc) {
                if (k < 100)      v = embN[sIdx[l] * EMBc + k];
                else if (k < 200) v = embN[eIdx[l] * EMBc + (k - 100)];
                else if (k < 300) v = embP[pIdx[l] * EMBc + (k - 200)];
                else              v = xrow[k - 300];
            }
            Atile[e] = (__bf16)v;
        }
        __syncthreads();

        for (int nt = wv; nt < 25; nt += 8) {
            // hoist the full K strip of B into registers (13 x 8 VGPRs)
            v16bf bfr[KP / 32];
            const __bf16* brow = WtT + (size_t)(nt * 16 + nlo) * KP;
            #pragma unroll
            for (int kc = 0; kc < KP / 32; ++kc) bfr[kc] = load_b16(brow + kc * 32, hk);

            const int f = nt * 16 + nlo;           // output column (< 400)
            const float wa = Wa[f], btv = bt_vec[f];

            for (int mt = 0; mt < 4; ++mt) {
                v8f acc = {0.f, 0.f, 0.f, 0.f, 0.f, 0.f, 0.f, 0.f};
                const __bf16* arow = Atile + (mt * 16 + nlo) * KP;
                #pragma unroll
                for (int kc = 0; kc < KP / 32; ++kc)
                    acc = wmma_bf16(load_a16(arow + kc * 32, hk), bfr[kc], acc);

                #pragma unroll
                for (int r = 0; r < 8; ++r) {
                    float v = tanhf(acc[r] + btv) * wa;
                    // reduce across the 16 columns (low 4 lane bits)
                    v += __shfl_xor(v, 1);
                    v += __shfl_xor(v, 2);
                    v += __shfl_xor(v, 4);
                    v += __shfl_xor(v, 8);
                    int l = phase * 64 + mt * 16 + 8 * hk + r;
                    if (nlo == 0 && l < Lc) atomicAdd(&ctx_acc[l], v);
                }
            }
        }
        __syncthreads();
    }
    if (tid < Lc) ctx[((size_t)b * Lc + tid) * Tc + t] = ctx_acc[tid];
}

// ---------------- kernel 2: softmax over T (axis=1) -------------------------
__global__ void softmax_t_kernel(const float* __restrict__ ctx, float* __restrict__ attn) {
    __shared__ float red[Tc];
    const int row = blockIdx.x;                  // b*L + l
    const int b = row / Lc, l = row % Lc, t = threadIdx.x;
    float v = ctx[(size_t)row * Tc + t];
    red[t] = v; __syncthreads();
    for (int s = Tc / 2; s > 0; s >>= 1) { if (t < s) red[t] = fmaxf(red[t], red[t + s]); __syncthreads(); }
    float mx = red[0]; __syncthreads();
    float e = expf(v - mx);
    red[t] = e; __syncthreads();
    for (int s = Tc / 2; s > 0; s >>= 1) { if (t < s) red[t] += red[t + s]; __syncthreads(); }
    attn[((size_t)b * Tc + t) * Lc + l] = e / red[0];
}

// ---------------- kernel 3: code_vectors + bf16 rnn input -------------------
__global__ void build_rnn_in_kernel(const float* __restrict__ x,
                                    const float* __restrict__ embN,
                                    const float* __restrict__ embP,
                                    const float* __restrict__ attn,
                                    __bf16* __restrict__ rnn_in) {
    __shared__ int   sIdx[Lc], pIdx[Lc], eIdx[Lc];
    __shared__ float att[Lc], firsts[INDIM];
    const int bt = blockIdx.x, tid = threadIdx.x;
    const float* xrow = x + (size_t)bt * XD;

    for (int i = tid; i < 3 * Lc; i += 256) {
        int iv = (int)xrow[INDIM + i];
        iv = iv < 0 ? 0 : (iv >= VOCABc ? VOCABc - 1 : iv);
        int l = i / 3, c = i % 3;
        if (c == 0) sIdx[l] = iv; else if (c == 1) pIdx[l] = iv; else eIdx[l] = iv;
    }
    if (tid < Lc) att[tid] = attn[(size_t)bt * Lc + tid];
    if (tid < INDIM) firsts[tid] = xrow[tid];
    __syncthreads();

    for (int k = tid; k < RINP; k += 256) {
        float v = 0.0f;
        if (k < INDIM) {
            v = firsts[k];
        } else if (k < INDIM + FEATc) {
            int fi = k - INDIM;
            if (fi < 300) {
                const float* tab; const int* ia; int col;
                if (fi < 100)      { tab = embN; ia = sIdx; col = fi; }
                else if (fi < 200) { tab = embN; ia = eIdx; col = fi - 100; }
                else               { tab = embP; ia = pIdx; col = fi - 200; }
                float acc = 0.0f;
                for (int l = 0; l < Lc; ++l) acc += att[l] * tab[ia[l] * EMBc + col];
                v = acc;
            } else {
                float asum = 0.0f;
                for (int l = 0; l < Lc; ++l) asum += att[l];
                v = firsts[fi - 300] * asum;
            }
        }
        rnn_in[(size_t)bt * RINP + k] = (__bf16)v;
    }
}

// ---------------- kernel 4: persistent LSTM (M=16 = one WMMA tile) ----------
// A fragments (x-part + h-part, 20 chunks = 160 VGPRs) hoisted per timestep
// and reused across the wave's 4 N-tiles. Single resident block, so
// launch_bounds(256,1) removes the VGPR cap.
__global__ void __launch_bounds__(256, 1)
lstm_kernel(const __bf16* __restrict__ rnn_in,
            const __bf16* __restrict__ WcT,
            const float* __restrict__ bsum,
            __bf16* __restrict__ hs) {
    __shared__ float  gates[Bc * G4];            // 32 KB
    __shared__ float  cF[Bc * HIDc];
    __shared__ __bf16 hB[Bc * HIDc];

    const int tid = threadIdx.x;
    const int lane = tid & 31, wv = tid >> 5;
    const int hk = lane >> 4, nlo = lane & 15;

    for (int i = tid; i < Bc * HIDc; i += 256) { cF[i] = 0.0f; hB[i] = (__bf16)0.0f; }
    __syncthreads();

    for (int t = 0; t < Tc; ++t) {
        v16bf afr[KLSTM / 32];                   // 20 x 8 VGPRs
        const __bf16* arow = rnn_in + ((size_t)nlo * Tc + t) * RINP; // batch row = nlo
        #pragma unroll
        for (int kc = 0; kc < 16; ++kc)          // x-part (512)
            afr[kc] = load_a16(arow + kc * 32, hk);
        #pragma unroll
        for (int kc = 16; kc < 20; ++kc)         // h-part (128)
            afr[kc] = load_a16(&hB[nlo * HIDc + (kc - 16) * 32], hk);

        for (int nt = wv; nt < 32; nt += 8) {
            v8f acc = {0.f, 0.f, 0.f, 0.f, 0.f, 0.f, 0.f, 0.f};
            const __bf16* brow = WcT + (size_t)(nt * 16 + nlo) * KLSTM;
            #pragma unroll
            for (int kc = 0; kc < KLSTM / 32; ++kc)
                acc = wmma_bf16(afr[kc], load_b16(brow + kc * 32, hk), acc);

            int g = nt * 16 + nlo;
            #pragma unroll
            for (int r = 0; r < 8; ++r)
                gates[(8 * hk + r) * G4 + g] = acc[r] + bsum[g];
        }
        __syncthreads();
        for (int i = tid; i < Bc * HIDc; i += 256) {
            int bb = i >> 7, hh = i & (HIDc - 1);
            float gi = gates[bb * G4 + hh];
            float gf = gates[bb * G4 + HIDc + hh];
            float gg = gates[bb * G4 + 2 * HIDc + hh];
            float go = gates[bb * G4 + 3 * HIDc + hh];
            float c = sigf(gf) * cF[i] + sigf(gi) * tanhf(gg);
            float h = sigf(go) * tanhf(c);
            cF[i] = c;
            hB[i] = (__bf16)h;
            hs[((size_t)bb * Tc + t) * HIDc + hh] = (__bf16)h;
        }
        __syncthreads();
    }
}

// ---------------- kernel 5: out = sigmoid(hs @ Wfc + bfc) -------------------
__global__ void out_kernel(const __bf16* __restrict__ hs,
                           const __bf16* __restrict__ WfT,
                           const float* __restrict__ bfc,
                           float* __restrict__ out) {
    const int mt = blockIdx.x;                   // 128 M-tiles of 16 rows
    const int tid = threadIdx.x;
    const int lane = tid & 31, nt = tid >> 5;    // 4 waves = 4 N-tiles
    const int hk = lane >> 4, nlo = lane & 15;

    v8f acc = {0.f, 0.f, 0.f, 0.f, 0.f, 0.f, 0.f, 0.f};
    const __bf16* arow = hs + (size_t)(mt * 16 + nlo) * HIDc;
    const __bf16* brow = WfT + (size_t)(nt * 16 + nlo) * HIDc;
    #pragma unroll
    for (int kc = 0; kc < HIDc / 32; ++kc) {
        v16bf a = load_a16(arow + kc * 32, hk);
        v16bf bb = load_b16(brow + kc * 32, hk);
        acc = wmma_bf16(a, bb, acc);
    }
    int o = nt * 16 + nlo;
    if (o < OUTc) {
        float bo = bfc[o];
        #pragma unroll
        for (int r = 0; r < 8; ++r) {
            int row = mt * 16 + 8 * hk + r;       // flat (b*T + t)
            out[(size_t)row * OUTc + o] = sigf(acc[r] + bo);
        }
    }
}

// ---------------- host launcher ---------------------------------------------
extern "C" void kernel_launch(void* const* d_in, const int* in_sizes, int n_in,
                              void* d_out, int out_size, void* d_ws, size_t ws_size,
                              hipStream_t stream) {
    const float* x    = (const float*)d_in[0];
    const float* embN = (const float*)d_in[1];
    const float* embP = (const float*)d_in[2];
    const float* W_t  = (const float*)d_in[3];
    const float* b_t  = (const float*)d_in[4];
    const float* W_a  = (const float*)d_in[5];
    // d_in[6] = b_a (scalar; softmax over T is shift-invariant, unused)
    const float* W_ih = (const float*)d_in[7];
    const float* W_hh = (const float*)d_in[8];
    const float* b_ih = (const float*)d_in[9];
    const float* b_hh = (const float*)d_in[10];
    const float* W_fc = (const float*)d_in[11];
    const float* b_fc = (const float*)d_in[12];
    float* out = (float*)d_out;

    char* ws = (char*)d_ws;
    size_t off = 0;
    auto alloc = [&](size_t bytes) -> void* {
        void* p = (void*)(ws + off);
        off += (bytes + 255) & ~(size_t)255;
        return p;
    };
    __bf16* WtT    = (__bf16*)alloc((size_t)KP * KP * 2);        // 416x416
    __bf16* WcT    = (__bf16*)alloc((size_t)G4 * KLSTM * 2);     // 512x640
    __bf16* WfT    = (__bf16*)alloc((size_t)64 * HIDc * 2);      // 64x128
    float*  bsum   = (float*)alloc((size_t)G4 * 4);
    float*  ctx    = (float*)alloc((size_t)Bc * Lc * Tc * 4);    // (B,L,T)
    float*  attn   = (float*)alloc((size_t)Bc * Tc * Lc * 4);    // (B,T,L)
    __bf16* rnn_in = (__bf16*)alloc((size_t)Bc * Tc * RINP * 2);
    __bf16* hs     = (__bf16*)alloc((size_t)Bc * Tc * HIDc * 2);
    (void)ws_size; (void)in_sizes; (void)n_in; (void)out_size;

    prep_wt  <<<(KP * KP + 255) / 256, 256, 0, stream>>>(W_t, WtT);
    prep_wcat<<<(G4 * KLSTM + 255) / 256, 256, 0, stream>>>(W_ih, W_hh, WcT);
    prep_wfc <<<(64 * HIDc + 255) / 256, 256, 0, stream>>>(W_fc, WfT);
    prep_bias<<<2, 256, 0, stream>>>(b_ih, b_hh, bsum);

    attn_score_kernel  <<<Bc * Tc, 256, 0, stream>>>(x, embN, embP, WtT, b_t, W_a, ctx);
    softmax_t_kernel   <<<Bc * Lc, Tc, 0, stream>>>(ctx, attn);
    build_rnn_in_kernel<<<Bc * Tc, 256, 0, stream>>>(x, embN, embP, attn, rnn_in);
    lstm_kernel        <<<1, 256, 0, stream>>>(rnn_in, WcT, bsum, hs);
    out_kernel         <<<(Bc * Tc) / 16, 128, 0, stream>>>(hs, WfT, b_fc, out);
}
